// MicroscopicEmbedding_850403525043
// MI455X (gfx1250) — matrix operand order repaired and checked by reading, not compiled, not run on hardware
//
#include <hip/hip_runtime.h>
#include <hip/hip_bf16.h>
#include <math.h>

typedef __attribute__((ext_vector_type(16))) _Float16     v16h;
typedef __attribute__((ext_vector_type(2)))  _Float16     h2;
typedef __attribute__((ext_vector_type(8)))  float        v8f;
typedef __attribute__((ext_vector_type(4)))  unsigned int v4u;
typedef __attribute__((ext_vector_type(8)))  unsigned int v8u;

#define MOE_ND     16
#define MOE_BASE   64
#define MOE_CSCALE 0.1f
#define MOE_WSTR   36   // LDS row stride in dwords (16B-aligned, bank-skewed)

// ---------------------------------------------------------------------------
// GELU via native gfx1250 v_tanh_f32 (TRANS op; v_nops cover the documented
// "1 op before result use" TRANS hazard). tanh-form == exact gelu to ~1e-7
// for the |x| <~ 1e-2 pre-activations this model produces.
// ---------------------------------------------------------------------------
static __device__ __forceinline__ float moe_gelu(float x) {
  const float u = 0.7978845608028654f * x * __builtin_fmaf(0.044715f, x * x, 1.0f);
  float th;
  asm("v_tanh_f32 %0, %1\n\tv_nop\n\tv_nop" : "=v"(th) : "v"(u));
  return 0.5f * x * (1.0f + th);
}

static __device__ __forceinline__ unsigned int moe_pk(float a, float b) {
  h2 p;
  p[0] = (_Float16)a;
  p[1] = (_Float16)b;
  return __builtin_bit_cast(unsigned int, p);
}

// ---------------------------------------------------------------------------
// A-fragment (16x32 f16, ISA 7.12.2): packed-f16 LDS rows, element e ->
// K = (e&8)*2 + half*8 + (e&7) (+32*kb). Pairs are K-consecutive, so one
// fragment = two 16B LDS loads per lane.
// ---------------------------------------------------------------------------
static __device__ __forceinline__ v16h moe_load_a_pk(const unsigned int* __restrict__ rowW,
                                                     int half, int woff) {
  const v4u lo = *(const v4u*)(rowW + woff + half * 4);
  const v4u hi = *(const v4u*)(rowW + woff + 8 + half * 4);
  v8u t;
#pragma unroll
  for (int i = 0; i < 4; ++i) { t[i] = lo[i]; t[i + 4] = hi[i]; }
  return __builtin_bit_cast(v16h, t);
}

static __device__ __forceinline__ v16h moe_load_a_pk_masked(const unsigned int* __restrict__ rowW,
                                                            int half, int woff, bool on) {
  const v4u zero = {0u, 0u, 0u, 0u};
  const v4u lo = on ? *(const v4u*)(rowW + woff + half * 4)     : zero;
  const v4u hi = on ? *(const v4u*)(rowW + woff + 8 + half * 4) : zero;
  v8u t;
#pragma unroll
  for (int i = 0; i < 4; ++i) { t[i] = lo[i]; t[i + 4] = hi[i]; }
  return __builtin_bit_cast(v16h, t);
}

// B-fragment from pre-swizzled f16 weights: one contiguous 32B load per lane.
// Fragment order: dom*256 + kb*128 + nt*32 + lane (in v16h units).
static __device__ __forceinline__ v8f moe_mm64_sw(const v16h a0, const v16h a1,
                                                  const v16h* __restrict__ Wv,
                                                  int dombase, int nt, int lane, v8f acc) {
  const v16h b0 = Wv[dombase + nt * 32 + lane];
  const v16h b1 = Wv[dombase + 128 + nt * 32 + lane];
  acc = __builtin_amdgcn_wmma_f32_16x16x32_f16(false, a0, false, b0, (short)0, acc, false, false);
  acc = __builtin_amdgcn_wmma_f32_16x16x32_f16(false, a1, false, b1, (short)0, acc, false, false);
  return acc;
}

// B-fragment straight from f32 global (no-workspace fallback path).
static __device__ __forceinline__ v16h moe_load_b(const float* __restrict__ W,
                                                  int half, int lane16, int n0, int koff) {
  v16h b;
#pragma unroll
  for (int e = 0; e < 16; ++e) {
    const int k = koff + (e & 8) * 2 + half * 8 + (e & 7);
    b[e] = (_Float16)W[k * 64 + n0 + lane16];
  }
  return b;
}

static __device__ __forceinline__ v8f moe_mm64(const v16h a0, const v16h a1,
                                               const float* __restrict__ W,
                                               int half, int lane16, int n0, v8f acc) {
  const v16h b0 = moe_load_b(W, half, lane16, n0, 0);
  const v16h b1 = moe_load_b(W, half, lane16, n0, 32);
  acc = __builtin_amdgcn_wmma_f32_16x16x32_f16(false, a0, false, b0, (short)0, acc, false, false);
  acc = __builtin_amdgcn_wmma_f32_16x16x32_f16(false, a1, false, b1, (short)0, acc, false, false);
  return acc;
}

// Pack a 16-row h tile (f32 global -> packed f16 LDS rows, stride 36 dwords).
static __device__ __forceinline__ void moe_pack_h(const float* __restrict__ hrow,
                                                  unsigned int* __restrict__ rowW, int half) {
  const float4* __restrict__ src4 = (const float4*)hrow;
#pragma unroll
  for (int j = 0; j < 4; ++j) {
    const float4 v0 = src4[half * 8 + j * 2];
    const float4 v1 = src4[half * 8 + j * 2 + 1];
    const v4u pk = { moe_pk(v0.x, v0.y), moe_pk(v0.z, v0.w),
                     moe_pk(v1.x, v1.y), moe_pk(v1.z, v1.w) };
    *(v4u*)(rowW + half * 16 + j * 4) = pk;
  }
}

// ---------------------------------------------------------------------------
// Binning pipeline: ws ints [0..15]=cnt [16..32]=offs [33..48]=cursor
//                   [64..64+ntok)=list ; then 256B-aligned W1s/W2s (f16 swizzled)
// ---------------------------------------------------------------------------
__global__ void moe_init_kernel(int* __restrict__ ws) {
  if (threadIdx.x < 64) ws[threadIdx.x] = 0;
}

__global__ void moe_count_kernel(const int* __restrict__ x, const int* __restrict__ tdom,
                                 int* __restrict__ ws, int ntok) {
  const int t = blockIdx.x * blockDim.x + threadIdx.x;
  if (t < ntok) {
    const int d = tdom[x[t]];
    if ((unsigned)d < (unsigned)MOE_ND) atomicAdd(&ws[d], 1);
  }
}

__global__ void moe_scan_kernel(int* __restrict__ ws) {
  if (threadIdx.x == 0) {
    int acc = 0;
#pragma unroll
    for (int i = 0; i < MOE_ND; ++i) {
      ws[16 + i] = acc;
      ws[33 + i] = acc;
      acc += ws[i];
    }
    ws[16 + MOE_ND] = acc;
  }
}

__global__ void moe_scatter_kernel(const int* __restrict__ x, const int* __restrict__ tdom,
                                   int* __restrict__ ws, int ntok) {
  const int t = blockIdx.x * blockDim.x + threadIdx.x;
  if (t < ntok) {
    const int d = tdom[x[t]];
    if ((unsigned)d < (unsigned)MOE_ND) {
      const int pos = atomicAdd(&ws[33 + d], 1);
      ws[64 + pos] = t;
    }
  }
}

// Swizzle W1/W2 f32 -> f16 in WMMA B-fragment lane order (one 32B blob/lane).
__global__ void moe_wswz_kernel(const float* __restrict__ W1, const float* __restrict__ W2,
                                _Float16* __restrict__ W1s, _Float16* __restrict__ W2s) {
  const int tid = blockIdx.x * blockDim.x + threadIdx.x;   // 8192 total
  if (tid >= 8192) return;
  const int sel  = tid >> 12;
  const int id   = tid & 4095;           // dom*256 + kb*128 + nt*32 + lane
  const int lane = id & 31;
  const int nt   = (id >> 5) & 3;
  const int kb   = (id >> 7) & 1;
  const int dom  = id >> 8;
  const int half = lane >> 4, lane16 = lane & 15;
  const float* __restrict__ W = (sel ? W2 : W1) + dom * 4096;
  v16h v;
#pragma unroll
  for (int e = 0; e < 16; ++e) {
    const int k = kb * 32 + (e & 8) * 2 + half * 8 + (e & 7);
    v[e] = (_Float16)W[k * 64 + nt * 16 + lane16];
  }
  *(v16h*)((sel ? W2s : W1s) + (size_t)id * 16) = v;
}

// out = base_embed[x] for UNASSIGNED tokens only (assigned rows are fully
// rewritten by moe_corr_kernel, so skip ~half the write traffic).
__global__ void moe_base_copy_kernel(const int* __restrict__ x, const int* __restrict__ tdom,
                                     const float4* __restrict__ be4,
                                     float4* __restrict__ out4, int ntok) {
  const int tid = blockIdx.x * blockDim.x + threadIdx.x;
  if (tid < ntok * 16) {
    const int tok = tid >> 4;
    const int j = tid & 15;
    const int xv = x[tok];
    if ((unsigned)tdom[xv] >= (unsigned)MOE_ND)
      out4[tid] = be4[(size_t)xv * 16 + j];
  }
}

// ---------------------------------------------------------------------------
// Main WMMA kernel: per-domain binned tokens, one 16-token tile per wave trip.
// ---------------------------------------------------------------------------
__global__ void __launch_bounds__(128) moe_corr_kernel(
    const float* __restrict__ base,
    const _Float16* __restrict__ W1s,
    const _Float16* __restrict__ W2s,
    const int* __restrict__ ws,
    float* __restrict__ out) {
  __shared__ __align__(16) unsigned int sHh[4][16 * MOE_WSTR];
  __shared__ __align__(16) _Float16     sHd[4][16 * MOE_WSTR * 2];

  const int lane   = threadIdx.x & 31;
  const int w      = threadIdx.x >> 5;
  const int lane16 = lane & 15;
  const int half   = lane >> 4;
  const int gwave  = blockIdx.x * 4 + w;
  const int nwave  = gridDim.x * 4;

  const int* offs = ws + 16;
  const int* list = ws + 64;
  const v16h* __restrict__ W1v = (const v16h*)W1s;
  const v16h* __restrict__ W2v = (const v16h*)W2s;

  unsigned int* hW = &sHh[w][0];
  _Float16* hidH = &sHd[w][0];
  const unsigned int* hidW = (const unsigned int*)hidH;

  for (int dom = 0; dom < MOE_ND; ++dom) {
    const int start = offs[dom];
    const int cnt   = offs[dom + 1] - start;
    if (cnt <= 0) continue;
    const int ntiles  = (cnt + 15) >> 4;
    const int dombase = dom * 256;

    if (gwave < ntiles) {   // pull this domain's 16KB of swizzled weights in
#pragma unroll
      for (int j = 0; j < 2; ++j) {
        __builtin_prefetch((const char*)(W1v + dombase) + (j * 32 + lane) * 128, 0, 1);
        __builtin_prefetch((const char*)(W2v + dombase) + (j * 32 + lane) * 128, 0, 1);
      }
    }

    for (int t = gwave; t < ntiles; t += nwave) {
      const int tbase = start + t * 16;
      const int rem   = cnt - t * 16;
      const int valid = rem < 16 ? rem : 16;

      // Gather 16 h rows -> packed f16 LDS (each lane: 8 b128 loads, 4 b128 stores)
      const int myTok = list[tbase + (lane16 < valid ? lane16 : valid - 1)];
      moe_pack_h(base + (size_t)myTok * MOE_BASE, hW + lane16 * MOE_WSTR, half);

      // ---- GEMM1: hid = gelu(h @ W1[dom]) ----
      const unsigned int* myRow = hW + lane16 * MOE_WSTR;
      const v16h a0 = moe_load_a_pk(myRow, half, 0);
      const v16h a1 = moe_load_a_pk(myRow, half, 16);
#pragma unroll
      for (int nt = 0; nt < 4; ++nt) {
        v8f acc = {};
        acc = moe_mm64_sw(a0, a1, W1v, dombase, nt, lane, acc);
#pragma unroll
        for (int r = 0; r < 8; ++r)
          hidH[(r + half * 8) * (MOE_WSTR * 2) + nt * 16 + lane16] = (_Float16)moe_gelu(acc[r]);
      }

      // ---- GEMM2: corr = hid @ W2[dom]; out = h + 0.1*corr ----
      const unsigned int* myHid = hidW + lane16 * MOE_WSTR;
      const v16h c0 = moe_load_a_pk(myHid, half, 0);
      const v16h c1 = moe_load_a_pk(myHid, half, 16);
#pragma unroll
      for (int nt = 0; nt < 4; ++nt) {
        const int n0 = nt * 16;
        v8f acc = {};
        acc = moe_mm64_sw(c0, c1, W2v, dombase, nt, lane, acc);
#pragma unroll
        for (int r = 0; r < 8; ++r) {
          const int m = r + half * 8;
          if (m < valid) {
            const int tok = list[tbase + m];
            out[(size_t)tok * MOE_BASE + n0 + lane16] =
                base[(size_t)tok * MOE_BASE + n0 + lane16] + MOE_CSCALE * acc[r];
          }
        }
      }
    }
  }
}

// ---------------------------------------------------------------------------
// Fallback (no workspace): masked all-domain accumulation; zeroed A rows give
// exactly-zero corrections (gelu(0)==0).
// ---------------------------------------------------------------------------
__global__ void __launch_bounds__(128) moe_dense_kernel(
    const int* __restrict__ x,
    const int* __restrict__ tdom,
    const float* __restrict__ base,
    const float* __restrict__ W1,
    const float* __restrict__ W2,
    float* __restrict__ out, int ntok) {
  __shared__ __align__(16) unsigned int sHh[4][16 * MOE_WSTR];
  __shared__ __align__(16) _Float16     sHd[4][16 * MOE_WSTR * 2];

  const int lane   = threadIdx.x & 31;
  const int w      = threadIdx.x >> 5;
  const int lane16 = lane & 15;
  const int half   = lane >> 4;
  const int gwave  = blockIdx.x * 4 + w;
  const int nwave  = gridDim.x * 4;

  unsigned int* hW = &sHh[w][0];
  _Float16* hidH = &sHd[w][0];
  const unsigned int* hidW = (const unsigned int*)hidH;

  const int ntiles = (ntok + 15) >> 4;
  for (int t = gwave; t < ntiles; t += nwave) {
    const int tbase = t * 16;
    const int rem   = ntok - tbase;
    const int valid = rem < 16 ? rem : 16;

    const int myTok = tbase + (lane16 < valid ? lane16 : valid - 1);
    const int myDom = tdom[x[myTok]];
    moe_pack_h(base + (size_t)myTok * MOE_BASE, hW + lane16 * MOE_WSTR, half);

    const unsigned int* myRow = hW + lane16 * MOE_WSTR;
    const unsigned int* myHid = hidW + lane16 * MOE_WSTR;

    v8f tt[4] = {};
    for (int dom = 0; dom < MOE_ND; ++dom) {
      const bool on = (myDom == dom);
      const v16h a0 = moe_load_a_pk_masked(myRow, half, 0, on);
      const v16h a1 = moe_load_a_pk_masked(myRow, half, 16, on);
      const float* W1d = W1 + dom * 4096;
      const float* W2d = W2 + dom * 4096;
#pragma unroll
      for (int nt = 0; nt < 4; ++nt) {
        v8f acc = {};
        acc = moe_mm64(a0, a1, W1d, half, lane16, nt * 16, acc);
#pragma unroll
        for (int r = 0; r < 8; ++r)
          hidH[(r + half * 8) * (MOE_WSTR * 2) + nt * 16 + lane16] = (_Float16)moe_gelu(acc[r]);
      }
      const v16h c0 = moe_load_a_pk(myHid, half, 0);
      const v16h c1 = moe_load_a_pk(myHid, half, 16);
#pragma unroll
      for (int nt = 0; nt < 4; ++nt)
        tt[nt] = moe_mm64(c0, c1, W2d, half, lane16, nt * 16, tt[nt]);
    }

#pragma unroll
    for (int nt = 0; nt < 4; ++nt) {
      const int n0 = nt * 16;
#pragma unroll
      for (int r = 0; r < 8; ++r) {
        const int m = r + half * 8;
        if (m < valid) {
          out[(size_t)(tbase + m) * MOE_BASE + n0 + lane16] =
              base[(size_t)(tbase + m) * MOE_BASE + n0 + lane16] + MOE_CSCALE * tt[nt][r];
        }
      }
    }
  }
}

// ---------------------------------------------------------------------------
extern "C" void kernel_launch(void* const* d_in, const int* in_sizes, int n_in,
                              void* d_out, int out_size, void* d_ws, size_t ws_size,
                              hipStream_t stream) {
  (void)n_in; (void)out_size;
  const int*   x    = (const int*)d_in[0];
  const int*   tdom = (const int*)d_in[1];
  const float* base = (const float*)d_in[2];
  const float* W1   = (const float*)d_in[3];
  const float* W2   = (const float*)d_in[4];
  float* out = (float*)d_out;
  const int ntok = in_sizes[0];   // B*S

  const size_t listBytes = (((size_t)(64 + ntok) * sizeof(int)) + 255) & ~(size_t)255;
  const size_t wBytes = (size_t)2 * 16 * 4096 * sizeof(_Float16);   // W1s + W2s
  const size_t need = listBytes + wBytes;

  if (d_ws != nullptr && ws_size >= need) {
    int* ws = (int*)d_ws;
    _Float16* W1s = (_Float16*)((char*)d_ws + listBytes);
    _Float16* W2s = W1s + 16 * 4096;
    const int blk = 256;
    moe_init_kernel<<<1, 64, 0, stream>>>(ws);
    moe_count_kernel<<<(ntok + blk - 1) / blk, blk, 0, stream>>>(x, tdom, ws, ntok);
    moe_scan_kernel<<<1, 32, 0, stream>>>(ws);
    moe_scatter_kernel<<<(ntok + blk - 1) / blk, blk, 0, stream>>>(x, tdom, ws, ntok);
    moe_wswz_kernel<<<32, 256, 0, stream>>>(W1, W2, W1s, W2s);
    const int nvec = ntok * 16;
    moe_base_copy_kernel<<<(nvec + blk - 1) / blk, blk, 0, stream>>>(
        x, tdom, (const float4*)base, (float4*)out, ntok);
    moe_corr_kernel<<<1024, 128, 0, stream>>>(base, W1s, W2s, ws, out);
  } else {
    const int ntiles = (ntok + 15) >> 4;
    int blocks = (ntiles + 3) / 4;
    if (blocks > 2048) blocks = 2048;
    if (blocks < 1) blocks = 1;
    moe_dense_kernel<<<blocks, 128, 0, stream>>>(x, tdom, base, W1, W2, out, ntok);
  }
}
